// Translator_68109591380667
// MI455X (gfx1250) — compile-verified, matrix operand order checked by reading, not run on hardware
//
#include <hip/hip_runtime.h>
#include <stdint.h>

#define IOU_THR 0.5f

typedef int   v8i __attribute__((ext_vector_type(8)));
typedef float v4f __attribute__((ext_vector_type(4)));

// ---------------- CDNA5 async global->LDS helpers (ASYNCcnt path) ----------------
__device__ __forceinline__ uint32_t lds_off_of(const void* p) {
  // Low 32 bits of a generic pointer into LDS == LDS byte offset.
  return (uint32_t)(uintptr_t)p;
}
__device__ __forceinline__ void async_copy_b128(const void* g, uint32_t lds_off) {
  uint64_t ga = (uint64_t)g;
  asm volatile("global_load_async_to_lds_b128 %0, %1, off"
               :: "v"(lds_off), "v"(ga) : "memory");
}
__device__ __forceinline__ void async_copy_b32(const void* g, uint32_t lds_off) {
  uint64_t ga = (uint64_t)g;
  asm volatile("global_load_async_to_lds_b32 %0, %1, off"
               :: "v"(lds_off), "v"(ga) : "memory");
}
__device__ __forceinline__ void wait_async0() {
  asm volatile("s_wait_asynccnt 0" ::: "memory");
}

// ---------------- K1: cxcywh -> ltrb ----------------
__global__ void k_prep_ltrb(const float4* __restrict__ boxes, float4* __restrict__ ltrb, int N) {
  int i = blockIdx.x * blockDim.x + threadIdx.x;
  if (i < N) {
    float4 b = boxes[i];                 // (cx, cy, w, h)
    ltrb[i] = make_float4(b.x - 0.5f * b.z, b.y - 0.5f * b.w,
                          b.x + 0.5f * b.z, b.y + 0.5f * b.w);
  }
}

// ---------------- K2: stable argsort(-scores) via WMMA rank reduction ----------------
// rank_i = #{k : s_k > s_i} + #{k < i : s_k == s_i}  (== stable argsort of -scores)
// Each wave32 owns 16 i's. Per 64-k step, a 16x64 IU8 indicator matrix A (built per the
// documented 8-bit A-matrix VGPR striping) is multiplied by an all-ones 64x16 B on the
// matrix pipe: D[m][n] += sum_k A[m][k].  Every column of the I32 C/D tile carries the ranks.
// Indicators are branch-free (bitwise | / & on bools) so the A-build is straight-line
// v_cmp + mask-combine + v_cndmask, not EXEC-mask gymnastics.
__device__ __forceinline__ uint32_t ind_full(float sk, float si, int kk, int i) {
  bool b = (sk > si) | ((sk == si) & (kk < i));
  return (uint32_t)b;
}
__device__ __forceinline__ uint32_t ind_tail(float sk, float si, int kk, int i, int N) {
  bool b = ((sk > si) | ((sk == si) & (kk < i))) & (kk < N);
  return (uint32_t)b;
}

__global__ void k_rank_scatter(const float* __restrict__ scores, int* __restrict__ order, int N) {
  __shared__ float ss[256];
  int t    = threadIdx.x;
  int lane = t & 31;
  int wave = t >> 5;
  int i_base = blockIdx.x * 128 + wave * 16;     // 8 waves x 16 rows = 128 i's per block
  int m    = lane & 15;
  int hi   = (lane >= 16) ? 8 : 0;               // K-half selection per A-matrix lane striping
  int i    = i_base + m;

  float si = scores[min(i, N - 1)];

  v8i acc  = {0, 0, 0, 0, 0, 0, 0, 0};
  v8i ones = {0x01010101, 0x01010101, 0x01010101, 0x01010101,
              0x01010101, 0x01010101, 0x01010101, 0x01010101};

  int fullTiles = N / 256;                       // tiles with no k-bounds check needed
  int ntiles    = (N + 255) / 256;

  for (int tile = 0; tile < ntiles; ++tile) {
    __syncthreads();                                       // prior tile reads done
    int k = tile * 256 + t;
    async_copy_b32(&scores[min(k, N - 1)], lds_off_of(&ss[t]));
    wait_async0();
    __syncthreads();                                       // all waves staged

    bool full = (tile < fullTiles);
    #pragma unroll
    for (int sub = 0; sub < 4; ++sub) {                    // 4 x (16x16x64) per staged tile
      int k0s = tile * 256 + sub * 64;
      v8i a;
      #pragma unroll
      for (int v = 0; v < 8; ++v) {
        // 8-bit A 16x64 layout: VGPR v holds K = (v>>1)*16 + hi + (v&1)*4 .. +3
        int kb = ((v >> 1) * 16) + hi + ((v & 1) * 4);
        const float4 s4 = *(const float4*)&ss[sub * 64 + kb]; // LDS b128, broadcast per half-wave
        int kk = k0s + kb;
        uint32_t r;
        if (full) {
          r = (ind_full(s4.x, si, kk + 0, i))
            | (ind_full(s4.y, si, kk + 1, i) << 8)
            | (ind_full(s4.z, si, kk + 2, i) << 16)
            | (ind_full(s4.w, si, kk + 3, i) << 24);
        } else {
          r = (ind_tail(s4.x, si, kk + 0, i, N))
            | (ind_tail(s4.y, si, kk + 1, i, N) << 8)
            | (ind_tail(s4.z, si, kk + 2, i, N) << 16)
            | (ind_tail(s4.w, si, kk + 3, i, N) << 24);
        }
        a[v] = (int)r;
      }
      // uniform control flow here: EXEC is all-ones as WMMA requires
      acc = __builtin_amdgcn_wmma_i32_16x16x64_iu8(false, a, false, ones, acc, false, false);
    }
  }

  // C/D layout: lane 0 VGPRs 0..7 = rows 0..7 (col 0); lane 16 VGPRs 0..7 = rows 8..15.
  if (lane == 0 || lane == 16) {
    int mbase = (lane == 0) ? 0 : 8;
    #pragma unroll
    for (int r = 0; r < 8; ++r) {
      int ii = i_base + mbase + r;
      if (ii < N) order[acc[r]] = ii;                      // ranks form a permutation
    }
  }
}

// ---------------- K3: gather boxes into sorted order ----------------
__global__ void k_gather_sorted(const float4* __restrict__ ltrb, const int* __restrict__ order,
                                float4* __restrict__ sltrb, int N) {
  int s = blockIdx.x * blockDim.x + threadIdx.x;
  if (s < N) sltrb[s] = ltrb[order[s]];
}

// ---------------- K4: NMS suppression bitmask (reference: UNCLAMPED inter) ----------------
__global__ void k_nms_mask(const float4* __restrict__ sltrb, uint32_t* __restrict__ mask,
                           int N, int padW) {
  __shared__ float4 cols[256];
  __shared__ float4 rows[32];
  int t  = threadIdx.x;
  int r  = t >> 3;                    // row within 32-row tile
  int wq = t & 7;                     // word within 8-word tile
  int w  = blockIdx.x * 8 + wq;       // global 32-col word
  int i  = blockIdx.y * 32 + r;       // sorted row
  int c  = blockIdx.x * 256 + t;      // staged column

  async_copy_b128(&sltrb[min(c, N - 1)], lds_off_of(&cols[t]));
  if (t < 32) {
    int ri = blockIdx.y * 32 + t;
    async_copy_b128(&sltrb[min(ri, N - 1)], lds_off_of(&rows[t]));
  }
  wait_async0();
  __syncthreads();

  float4 bi = rows[r];
  float area_i = (bi.z - bi.x) * (bi.w - bi.y);
  uint32_t word = 0;
  #pragma unroll 8
  for (int b = 0; b < 32; ++b) {
    int j = w * 32 + b;
    float4 bj = cols[wq * 32 + b];
    float iw = fminf(bi.z, bj.z) - fmaxf(bi.x, bj.x);   // no clamp (matches reference NMS)
    float ih = fminf(bi.w, bj.w) - fmaxf(bi.y, bj.y);
    float inter = iw * ih;
    float area_j = (bj.z - bj.x) * (bj.w - bj.y);
    float iou = inter / (area_i + area_j - inter);
    bool bit = (iou >= IOU_THR) & (j > i) & (j < N);    // branch-free
    word |= ((uint32_t)bit) << b;
  }
  if (i < N) mask[(size_t)i * padW + w] = word;
}

// ---------------- K5: sequential greedy scan (single wave32) + scatter keep ----------------
__global__ void k_nms_scan(const uint32_t* __restrict__ mask, const int* __restrict__ order,
                           float* __restrict__ keep_out, int N, int padW, int words) {
  __shared__ uint32_t keep[512];                       // padW <= 512
  int lane = threadIdx.x;                              // 0..31, single wave
  for (int w = lane; w < padW; w += 32) {
    int lo = w * 32;
    uint32_t v;
    if (lo + 32 <= N)      v = 0xFFFFFFFFu;
    else if (lo >= N)      v = 0u;
    else                   v = 0xFFFFFFFFu >> (32 - (N - lo));
    keep[w] = v;
  }
  __syncthreads();                                     // single-wave: S_NOP + compiler fence
  for (int cidx = 0; cidx < words; ++cidx) {
    uint32_t rem = keep[cidx];                         // uniform across lanes
    while (rem) {
      int b = __ffs(rem) - 1;
      int i = cidx * 32 + b;                           // kept row -> suppress its overlaps
      const uint32_t* mrow = mask + (size_t)i * padW;
      for (int w = lane; w < padW; w += 32)            // coalesced, each word owned by 1 lane
        keep[w] &= ~mrow[w];
      __syncthreads();
      uint32_t above = (b == 31) ? 0u : (0xFFFFFFFFu << (b + 1));
      rem = keep[cidx] & above;                        // re-read: in-chunk suppressions applied
    }
    __syncthreads();
  }
  for (int s = lane; s < N; s += 32) {                 // scatter back to original order
    uint32_t bit = (keep[s >> 5] >> (s & 31)) & 1u;
    keep_out[order[s]] = bit ? 1.0f : 0.0f;
  }
}

// ---------------- K6: full pairwise IoU (4-wide register blocking, b128 NT streams) ------
__device__ __forceinline__ float iou_clamped(float4 bi, float area_i, float4 bj, float area_j) {
  float iw = fmaxf(fminf(bi.z, bj.z) - fmaxf(bi.x, bj.x), 0.0f);
  float ih = fmaxf(fminf(bi.w, bj.w) - fmaxf(bi.y, bj.y), 0.0f);
  float inter = iw * ih;
  return inter / (area_i + area_j - inter);
}

__global__ void k_iou_full(const float4* __restrict__ ltrb, float* __restrict__ out, int N) {
  __shared__ float4 rows[128];
  int t  = threadIdx.x;
  int j0 = (blockIdx.x * 256 + t) * 4;                 // 4 columns per thread
  int i0 = blockIdx.y * 128;
  if (t < 128) async_copy_b128(&ltrb[min(i0 + t, N - 1)], lds_off_of(&rows[t]));
  wait_async0();
  __syncthreads();                                     // no barriers after this point
  if (j0 >= N) return;

  bool vec = ((N & 3) == 0) && (j0 + 3 < N);           // b128 stores need 16B alignment: N%4==0
  float4 b0 = ltrb[j0];
  float4 b1 = ltrb[min(j0 + 1, N - 1)];
  float4 b2 = ltrb[min(j0 + 2, N - 1)];
  float4 b3 = ltrb[min(j0 + 3, N - 1)];
  float a0 = (b0.z - b0.x) * (b0.w - b0.y);
  float a1 = (b1.z - b1.x) * (b1.w - b1.y);
  float a2 = (b2.z - b2.x) * (b2.w - b2.y);
  float a3 = (b3.z - b3.x) * (b3.w - b3.y);

  int rmax = min(128, N - i0);
  for (int rr = 0; rr < rmax; ++rr) {
    float4 bi = rows[rr];
    float area_i = (bi.z - bi.x) * (bi.w - bi.y);
    v4f res;
    res.x = iou_clamped(bi, area_i, b0, a0);
    res.y = iou_clamped(bi, area_i, b1, a1);
    res.z = iou_clamped(bi, area_i, b2, a2);
    res.w = iou_clamped(bi, area_i, b3, a3);
    size_t base = (size_t)(i0 + rr) * N + j0;
    if (vec) {
      __builtin_nontemporal_store(res, (v4f*)&out[base]);     // 16B NT stream
    } else {
      if (j0 + 0 < N) __builtin_nontemporal_store(res.x, &out[base + 0]);
      if (j0 + 1 < N) __builtin_nontemporal_store(res.y, &out[base + 1]);
      if (j0 + 2 < N) __builtin_nontemporal_store(res.z, &out[base + 2]);
      if (j0 + 3 < N) __builtin_nontemporal_store(res.w, &out[base + 3]);
    }
  }
}

extern "C" void kernel_launch(void* const* d_in, const int* in_sizes, int n_in,
                              void* d_out, int out_size, void* d_ws, size_t ws_size,
                              hipStream_t stream) {
  const float4* boxes  = (const float4*)d_in[0];   // N x 4 cxcywh
  const float*  scores = (const float*)d_in[1];    // N
  int N = in_sizes[1];

  float* out_iou  = (float*)d_out;                        // N*N
  float* out_keep = (float*)d_out + (size_t)N * N;        // N

  // small workspace: ltrb, sorted ltrb, order  (~360 KB for N=10000)
  float4* ltrb  = (float4*)d_ws;
  float4* sltrb = ltrb + N;
  int*    order = (int*)(sltrb + N);

  // 12.8MB suppression mask lives in d_out's IoU region; consumed by k_nms_scan
  // before k_iou_full overwrites it (stream-ordered).
  uint32_t* mask = (uint32_t*)d_out;

  int words = (N + 31) / 32;
  int padW  = ((words + 31) / 32) * 32;            // multiple of 32, <= 512 for N <= 16384

  int blocks256 = (N + 255) / 256;
  k_prep_ltrb   <<<blocks256, 256, 0, stream>>>(boxes, ltrb, N);

  int blocksRank = (N + 127) / 128;                // 128 i's per 8-wave block
  k_rank_scatter<<<blocksRank, 256, 0, stream>>>(scores, order, N);

  k_gather_sorted<<<blocks256, 256, 0, stream>>>(ltrb, order, sltrb, N);

  dim3 gmask((unsigned)(padW / 8), (unsigned)((N + 31) / 32));
  k_nms_mask<<<gmask, 256, 0, stream>>>(sltrb, mask, N, padW);

  k_nms_scan<<<1, 32, 0, stream>>>(mask, order, out_keep, N, padW, words);

  dim3 giou((unsigned)((N + 1023) / 1024), (unsigned)((N + 127) / 128));
  k_iou_full<<<giou, 256, 0, stream>>>(ltrb, out_iou, N);
}
